// NetworkRNN_80247168958997
// MI455X (gfx1250) — matrix operand order repaired and checked by reading, not compiled, not run on hardware
//
#include <hip/hip_runtime.h>
#include <math.h>

// ---------------- model constants ----------------
#define NPOP      2048
#define DT        0.1f
#define TAU_S     5.0f
#define TAU_E     10.0f
#define TAU_I     8.0f
#define V_REST    (-65.0f)
#define THETA     (-50.0f)
#define SIGMA     5.0f
#define E_EXC     0.0f
#define E_INH     (-70.0f)
#define TWO_PI_F  0.031415926535897934f   // 2*pi*IN_FREQ (IN_FREQ=0.005/ms)

#define A_S (DT / TAU_S)

// ---------------- launch geometry ----------------
#define SIM_BLOCKS   64
#define SIM_THREADS  256
#define WAVES_PER_BLOCK (SIM_THREADS / 32)   // 8 -> 512 waves total
#define NMAT 4          // EE, EI, IE, II (inp matvec collapses to colsum)
#define NT   128        // 2048/16 n-tiles per matrix
#define KT   64         // 2048/32 k-tiles
#define TILE_HALFS 512  // 32(K) x 16(N) bf16 per tile

typedef __attribute__((ext_vector_type(16))) __bf16 bf16x16;
typedef __attribute__((ext_vector_type(8)))  float  floatx8;

__device__ __forceinline__ unsigned short f2bf(float x) {
    unsigned u = __builtin_bit_cast(unsigned, x);
    u += 0x7fffu + ((u >> 16) & 1u);          // round-to-nearest-even
    return (unsigned short)(u >> 16);
}

__device__ __forceinline__ float sigm(float x) {
    return 1.0f / (1.0f + __expf(-x));
}

// rate-vector A-operand swizzle: element j=kt*32+o of r lands at rA[kt*32 + grp*16 + h]
// (ISA 16-bit A 16x32 layout: lanes 0-15 hold K{0..7,16..23}; lanes 16-31 hold K{8..15,24..31})
__device__ __forceinline__ int rA_index(int j) {
    int kt = j >> 5, o = j & 31;
    int grp = (o >> 3) & 1;
    int h = (o & 7) + ((o >= 16) ? 8 : 0);
    return kt * 32 + grp * 16 + h;
}

// ---------------- one-time weight packing: f32 row-major -> bf16 WMMA-B tiles ----------------
// Packed layout: Bp[m][nt][kt][grp(2)][nl(16)][h(16)]  (h -> K = 8*grp + (h&7) + 16*(h>>3), ISA table order)
__global__ void pack_weights(const float* __restrict__ W_EE, const float* __restrict__ W_EI,
                             const float* __restrict__ W_IE, const float* __restrict__ W_II,
                             unsigned short* __restrict__ Bp) {
    size_t p = (size_t)blockIdx.x * blockDim.x + threadIdx.x;
    int h   = (int)(p & 15);
    int nl  = (int)((p >> 4) & 15);
    int grp = (int)((p >> 8) & 1);
    int kt  = (int)((p >> 9) & 63);
    int nt  = (int)((p >> 15) & 127);
    int m   = (int)(p >> 22);
    int k = kt * 32 + grp * 8 + (h & 7) + ((h >> 3) << 4);
    int n = nt * 16 + nl;
    const float* W = (m == 0) ? W_EE : (m == 1) ? W_EI : (m == 2) ? W_IE : W_II;
    Bp[p] = f2bf(W[(size_t)k * NPOP + n]);
}

// colsum of W_inp_E: since r_in is uniform, r_in @ W = r_in_scalar * colsum(W)
__global__ void colsum_kernel(const float* __restrict__ W, float* __restrict__ cs) {
    int j = blockIdx.x * blockDim.x + threadIdx.x;
    float s = 0.0f;
    for (int k = 0; k < NPOP; ++k) s += W[(size_t)k * NPOP + j];
    cs[j] = s;
}

__global__ void init_state(const float* __restrict__ v0E, const float* __restrict__ v0I,
                           const float* __restrict__ g0_inp, const float* __restrict__ g0_EE,
                           const float* __restrict__ g0_EI, const float* __restrict__ g0_IE,
                           const float* __restrict__ g0_II,
                           float* __restrict__ g,     // 5*NPOP: inp,EE,EI,IE,II
                           float* __restrict__ vE, float* __restrict__ vI,
                           unsigned short* __restrict__ rAE, unsigned short* __restrict__ rAI,
                           float* __restrict__ stab, int* __restrict__ bar) {
    int j = blockIdx.x * blockDim.x + threadIdx.x;
    vE[j] = v0E[j];
    vI[j] = v0I[j];
    g[0 * NPOP + j] = g0_inp[j];
    g[1 * NPOP + j] = g0_EE[j];
    g[2 * NPOP + j] = g0_EI[j];
    g[3 * NPOP + j] = g0_IE[j];
    g[4 * NPOP + j] = g0_II[j];
    float rE = sigm((v0E[j] - THETA) / SIGMA);
    float rI = sigm((v0I[j] - THETA) / SIGMA);
    int ai = rA_index(j);
    rAE[ai] = f2bf(rE);
    rAI[ai] = f2bf(rI);
    if (j == 0) { *stab = 0.0f; *bar = 0; }
}

// monotonic grid barrier (64 resident blocks)
__device__ __forceinline__ void grid_sync(int* bar, int target) {
    __syncthreads();
    if (threadIdx.x == 0) {
        __threadfence();
        __hip_atomic_fetch_add(bar, 1, __ATOMIC_RELEASE, __HIP_MEMORY_SCOPE_AGENT);
        while (__hip_atomic_load(bar, __ATOMIC_ACQUIRE, __HIP_MEMORY_SCOPE_AGENT) < target) {
            __builtin_amdgcn_s_sleep(1);
        }
    }
    __syncthreads();
}

__global__ void __launch_bounds__(SIM_THREADS, 1)
sim_kernel(const unsigned short* __restrict__ Bp, const float* __restrict__ colsum,
           float* __restrict__ g, float* __restrict__ vE, float* __restrict__ vI,
           float* __restrict__ y,                     // 4*NPOP: yEE,yEI,yIE,yII
           unsigned short* __restrict__ rAE, unsigned short* __restrict__ rAI,
           float* __restrict__ stab, int* __restrict__ bar,
           float* __restrict__ out, const int* __restrict__ Tp) {
    __shared__ float red[SIM_THREADS];
    const int T = *Tp;

    const int lane = threadIdx.x & 31;
    const int w    = blockIdx.x * WAVES_PER_BLOCK + (threadIdx.x >> 5);  // 0..511
    const int m    = w >> 7;                                             // 0..3
    const int nt   = w & (NT - 1);
    const int grp  = lane >> 4;
    const unsigned short* rA  = (m < 2) ? rAE : rAI;                     // EE,EI use r_E; IE,II use r_I
    const unsigned short* Bt  = Bp + (size_t)w * (KT * TILE_HALFS);
    const unsigned short* ap0 = rA + grp * 16;
    const unsigned short* bp0 = Bt + grp * 256 + (lane & 15) * 16;

    int sync = 0;
    for (int t = 0; t < T; ++t) {
        // ---- phase 1: y = r_pre @ W  (bf16 WMMA, f32 accumulate) ----
        {
            floatx8 acc = {0.f, 0.f, 0.f, 0.f, 0.f, 0.f, 0.f, 0.f};
            for (int kt = 0; kt < KT; ++kt) {
                bf16x16 a = *(const bf16x16*)(ap0 + kt * 32);
                bf16x16 b = *(const bf16x16*)(bp0 + kt * TILE_HALFS);
                __builtin_prefetch(bp0 + (kt + 6) * TILE_HALFS, 0, 0);
                acc = __builtin_amdgcn_wmma_f32_16x16x32_bf16(
                        false, a, false, b, (short)0, acc, false, false);
            }
            // C/D layout: VGPR0, lanes 0-15 => row M=0, N=lane (rows identical by construction)
            if (lane < 16) y[m * NPOP + nt * 16 + lane] = acc[0];
        }
        grid_sync(bar, (++sync) * SIM_BLOCKS);

        // ---- phase 2: synapse + membrane update, rates out (block 0 only) ----
        if (blockIdx.x == 0) {
            const float t_ms = (float)t * DT;
            const float r_in = 0.5f * (1.0f + sinf(TWO_PI_F * t_ms));
            float local = 0.0f;
            for (int i = 0; i < NPOP / SIM_THREADS; ++i) {
                int j = threadIdx.x + i * SIM_THREADS;
                float vEj = vE[j], vIj = vI[j];

                float gi  = g[0 * NPOP + j];  gi  += A_S * (-gi  + r_in * colsum[j]);
                float gee = g[1 * NPOP + j];  gee += A_S * (-gee + y[0 * NPOP + j]);
                float gei = g[2 * NPOP + j];  gei += A_S * (-gei + y[1 * NPOP + j]);
                float gie = g[3 * NPOP + j];  gie += A_S * (-gie + y[2 * NPOP + j]);
                float gii = g[4 * NPOP + j];  gii += A_S * (-gii + y[3 * NPOP + j]);

                float I_E = gi  * (E_EXC - vEj) + gee * (E_EXC - vEj) + gie * (E_INH - vEj);
                float I_I = gei * (E_EXC - vIj) + gii * (E_INH - vIj);

                float dvE = (DT / TAU_E) * (-(vEj - V_REST) + I_E);
                float dvI = (DT / TAU_I) * (-(vIj - V_REST) + I_I);
                vEj += dvE; vIj += dvI;
                local += dvE * dvE + dvI * dvI;

                float rE = sigm((vEj - THETA) / SIGMA);
                float rI = sigm((vIj - THETA) / SIGMA);

                g[0 * NPOP + j] = gi;  g[1 * NPOP + j] = gee; g[2 * NPOP + j] = gei;
                g[3 * NPOP + j] = gie; g[4 * NPOP + j] = gii;
                vE[j] = vEj; vI[j] = vIj;

                out[(size_t)t * (2 * NPOP) + j]        = rE;
                out[(size_t)t * (2 * NPOP) + NPOP + j] = rI;

                int ai = rA_index(j);
                rAE[ai] = f2bf(rE);
                rAI[ai] = f2bf(rI);
            }
            red[threadIdx.x] = local;
            __syncthreads();
            for (int s2 = SIM_THREADS / 2; s2 > 0; s2 >>= 1) {
                if (threadIdx.x < s2) red[threadIdx.x] += red[threadIdx.x + s2];
                __syncthreads();
            }
            if (threadIdx.x == 0) {
                *stab += red[0] * (1.0f / (float)NPOP);
                if (t == T - 1) out[(size_t)T * (2 * NPOP)] = *stab;
            }
        }
        grid_sync(bar, (++sync) * SIM_BLOCKS);
    }
}

extern "C" void kernel_launch(void* const* d_in, const int* in_sizes, int n_in,
                              void* d_out, int out_size, void* d_ws, size_t ws_size,
                              hipStream_t stream) {
    const float* W_inp = (const float*)d_in[0];
    const float* W_EE  = (const float*)d_in[1];
    const float* W_EI  = (const float*)d_in[2];
    const float* W_IE  = (const float*)d_in[3];
    const float* W_II  = (const float*)d_in[4];
    const float* v0E   = (const float*)d_in[5];
    const float* v0I   = (const float*)d_in[6];
    const float* g0_inp = (const float*)d_in[7];
    const float* g0_EE  = (const float*)d_in[8];
    const float* g0_EI  = (const float*)d_in[9];
    const float* g0_IE  = (const float*)d_in[10];
    const float* g0_II  = (const float*)d_in[11];
    const int*   Tp     = (const int*)d_in[12];
    float* out = (float*)d_out;

    char* ws = (char*)d_ws;
    size_t off = 0;
    unsigned short* Bp = (unsigned short*)(ws + off); off += (size_t)NMAT * NPOP * NPOP * 2; // 32 MB
    float* colsum = (float*)(ws + off); off += NPOP * 4;
    float* g      = (float*)(ws + off); off += 5 * NPOP * 4;
    float* vE     = (float*)(ws + off); off += NPOP * 4;
    float* vI     = (float*)(ws + off); off += NPOP * 4;
    float* y      = (float*)(ws + off); off += 4 * NPOP * 4;
    unsigned short* rAE = (unsigned short*)(ws + off); off += NPOP * 2;
    unsigned short* rAI = (unsigned short*)(ws + off); off += NPOP * 2;
    float* stab = (float*)(ws + off); off += 4;
    int*   bar  = (int*)(ws + off);   off += 4;

    // one-time prep (recomputed every call for determinism)
    pack_weights<<<(NMAT * NPOP * NPOP) / 256, 256, 0, stream>>>(W_EE, W_EI, W_IE, W_II, Bp);
    colsum_kernel<<<NPOP / 256, 256, 0, stream>>>(W_inp, colsum);
    init_state<<<NPOP / 256, 256, 0, stream>>>(v0E, v0I, g0_inp, g0_EE, g0_EI, g0_IE, g0_II,
                                               g, vE, vI, rAE, rAI, stab, bar);
    // persistent time-stepping kernel
    sim_kernel<<<SIM_BLOCKS, SIM_THREADS, 0, stream>>>(Bp, colsum, g, vE, vI, y,
                                                       rAE, rAI, stab, bar, out, Tp);
}
